// GptOssGroupedExperts_1005022347969
// MI455X (gfx1250) — compile-verified, matrix operand order checked by reading, not run on hardware
//
#include <hip/hip_runtime.h>
#include <hip/hip_bf16.h>

// ---------------- problem constants (fixed by setup_inputs) ----------------
constexpr int E    = 8;
constexpr int D    = 2880;     // model dim
constexpr int H    = 2880;     // hidden dim
constexpr int NTOK = 16384;    // tokens
constexpr int H2   = 2 * H;    // 5760 (interleaved glu/lin)

constexpr float ALPHA = 1.702f;
constexpr float LIMIT = 7.0f;

// ---------------- tile config ----------------
constexpr int BM = 256;   // tokens per block tile (2048 % 256 == 0)
constexpr int BN = 64;    // output columns per block tile (5760/64=90, 2880/64=45)
constexpr int BK = 32;    // WMMA bf16 K depth

constexpr int LDS_ROW = 80;             // 64B of data + 16B skew -> conflict-free b128 reads
constexpr int A_TILE_BYTES = BM * LDS_ROW;   // 20480
constexpr int B_TILE_BYTES = BN * LDS_ROW;   //  5120

typedef __attribute__((ext_vector_type(16))) __bf16 v16bf;
typedef __attribute__((ext_vector_type(8)))  __bf16 v8bf;
typedef __attribute__((ext_vector_type(8)))  float  v8f;
typedef __attribute__((ext_vector_type(4)))  int    v4i;

#define AS_GLOBAL __attribute__((address_space(1)))
#define AS_LOCAL  __attribute__((address_space(3)))

#if __has_builtin(__builtin_amdgcn_global_load_async_to_lds_b128) && \
    __has_builtin(__builtin_amdgcn_s_wait_asynccnt)
#define HAS_ASYNC_LDS 1
#else
#define HAS_ASYNC_LDS 0
#endif

// 16-byte global->LDS copy: async DMA-style on CDNA5, sync fallback otherwise.
__device__ __forceinline__ void cp16_g2l(const __bf16* __restrict__ g, char* l) {
#if HAS_ASYNC_LDS
    __builtin_amdgcn_global_load_async_to_lds_b128(
        (AS_GLOBAL v4i*)g, (AS_LOCAL v4i*)l, /*offset=*/0, /*cpol=*/0);
#else
    *(v8bf*)l = *(const v8bf*)g;
#endif
}

template <int N>
__device__ __forceinline__ void wait_async() {
#if HAS_ASYNC_LDS
    __builtin_amdgcn_s_wait_asynccnt(N);
#endif
}

// ---------------- fp32 -> bf16 conversion (8 elems/thread) ----------------
__global__ __launch_bounds__(256)
void cvt_f32_bf16_kernel(const float* __restrict__ src, __bf16* __restrict__ dst, long n) {
    long i = ((long)blockIdx.x * 256 + threadIdx.x) * 8;
    if (i >= n) return;
    if (i + 8 <= n) {
        const float4 f0 = *(const float4*)(src + i);
        const float4 f1 = *(const float4*)(src + i + 4);
        v8bf o;
        o[0] = (__bf16)f0.x; o[1] = (__bf16)f0.y; o[2] = (__bf16)f0.z; o[3] = (__bf16)f0.w;
        o[4] = (__bf16)f1.x; o[5] = (__bf16)f1.y; o[6] = (__bf16)f1.z; o[7] = (__bf16)f1.w;
        *(v8bf*)(dst + i) = o;
    } else {
        for (long j = i; j < n; ++j) dst[j] = (__bf16)src[j];
    }
}

// ---------------- expert lookup ----------------
__device__ __forceinline__ int expert_of(const int* __restrict__ ntpe, int r0,
                                         int& estart, int& eend) {
    int e = -1; estart = 0; eend = 0;
    int cum = 0;
#pragma unroll
    for (int i = 0; i < E; ++i) {
        int c = ntpe[i];
        if (e < 0 && r0 < cum + c) { e = i; estart = cum; eend = cum + c; }
        cum += c;
    }
    return e;
}

// ---------------- staging: one BK-slice of A (BMxBK) and B (BNxBK) into LDS ----------------
// A chunks: 256 rows * 64B = 1024 x 16B -> 4 per thread. B: 64 rows -> 1 per thread.
__device__ __forceinline__ void issue_tile(const __bf16* __restrict__ Ag,
                                           const __bf16* __restrict__ Bg,
                                           int K, int k0, char* abuf, char* bbuf) {
    const int t = threadIdx.x;
#pragma unroll
    for (int j = 0; j < 4; ++j) {
        const int c = j * 256 + t;
        const int row = c >> 2, sub = c & 3;
        cp16_g2l(Ag + (size_t)row * K + k0 + sub * 8, abuf + row * LDS_ROW + sub * 16);
    }
    {
        const int row = t >> 2, sub = t & 3;
        cp16_g2l(Bg + (size_t)row * K + k0 + sub * 8, bbuf + row * LDS_ROW + sub * 16);
    }
}

// 32B WMMA fragment from LDS as two 16B loads (rows are only 16B aligned due to skew)
__device__ __forceinline__ v16bf frag_lds(const char* __restrict__ p) {
    const v8bf lo = *(const v8bf*)(p);
    const v8bf hi = *(const v8bf*)(p + 16);
    return __builtin_shufflevector(lo, hi, 0, 1, 2, 3, 4, 5, 6, 7,
                                           8, 9, 10, 11, 12, 13, 14, 15);
}

// 4x2 WMMA tiles per wave from the current LDS buffers
__device__ __forceinline__ void compute_step(const char* __restrict__ abuf,
                                             const char* __restrict__ bbuf,
                                             int wr, int wc, int lane, v8f acc[4][2]) {
    const int kb = (lane >> 4) * 32;   // byte offset of this lane's K-half
    const int rl = lane & 15;
    v16bf a[4], b[2];
#pragma unroll
    for (int mi = 0; mi < 4; ++mi)
        a[mi] = frag_lds(abuf + (wr * 64 + mi * 16 + rl) * LDS_ROW + kb);
#pragma unroll
    for (int ni = 0; ni < 2; ++ni)
        b[ni] = frag_lds(bbuf + (wc * 32 + ni * 16 + rl) * LDS_ROW + kb);
#pragma unroll
    for (int mi = 0; mi < 4; ++mi)
#pragma unroll
        for (int ni = 0; ni < 2; ++ni)
            acc[mi][ni] = __builtin_amdgcn_wmma_f32_16x16x32_bf16(
                false, a[mi], false, b[ni], (short)0, acc[mi][ni], false, false);
}

// double-buffered async-staged K loop; Ag/Bg are block-level row bases, row stride == K
__device__ __forceinline__ void staged_mma(const __bf16* __restrict__ Ag,
                                           const __bf16* __restrict__ Bg,
                                           int K, v8f acc[4][2]) {
    __shared__ __align__(128) char lds_a[2][A_TILE_BYTES];
    __shared__ __align__(128) char lds_b[2][B_TILE_BYTES];
    const int lane = threadIdx.x & 31;
    const int wave = threadIdx.x >> 5;
    const int wr = wave >> 1;   // 0..3 : 64-row slice
    const int wc = wave & 1;    // 0..1 : 32-col slice
    const int iters = K / BK;

    issue_tile(Ag, Bg, K, 0,  lds_a[0], lds_b[0]);
    issue_tile(Ag, Bg, K, BK, lds_a[1], lds_b[1]);

    for (int it = 0; it < iters - 1; ++it) {
        const int cur = it & 1;
        wait_async<5>();            // current buffer's 5 chunks landed (next 5 may fly)
        __syncthreads();
        compute_step(lds_a[cur], lds_b[cur], wr, wc, lane, acc);
        __syncthreads();            // everyone done reading buf[cur]
        if (it + 2 < iters)
            issue_tile(Ag, Bg, K, (it + 2) * BK, lds_a[cur], lds_b[cur]);
    }
    wait_async<0>();
    __syncthreads();
    compute_step(lds_a[(iters - 1) & 1], lds_b[(iters - 1) & 1], wr, wc, lane, acc);
}

// ---------------- GEMM1 + bias + swiglu, writes bf16 h[N,H] ----------------
__global__ __launch_bounds__(256)
void gemm1_swiglu_kernel(const __bf16* __restrict__ xb,
                         const __bf16* __restrict__ w1b,
                         const float*  __restrict__ bias1,
                         const int*    __restrict__ ntpe,
                         __bf16* __restrict__ hb) {
    const int lane = threadIdx.x & 31;
    const int wave = threadIdx.x >> 5;
    const int wr = wave >> 1, wc = wave & 1;
    const int r0 = blockIdx.y * BM;
    const int c0 = blockIdx.x * BN;

    int estart, eend;
    const int e = expert_of(ntpe, r0, estart, eend);
    if (e < 0) return;  // rows past total: GEMM2 zeroes these rows of d_out

    const __bf16* Ag = xb  + (size_t)r0 * D;
    const __bf16* Bg = w1b + ((size_t)e * H2 + c0) * (size_t)D;

    v8f acc[4][2];
#pragma unroll
    for (int mi = 0; mi < 4; ++mi)
#pragma unroll
        for (int ni = 0; ni < 2; ++ni) acc[mi][ni] = 0.0f;

    staged_mma(Ag, Bg, D, acc);

    // epilogue: bias + swiglu (adjacent-lane pairing) + bf16 store
#pragma unroll
    for (int ni = 0; ni < 2; ++ni) {
        const int col = c0 + wc * 32 + ni * 16 + (lane & 15);   // in [0, 2H)
        const float bv = bias1[(size_t)e * H2 + col];
#pragma unroll
        for (int mi = 0; mi < 4; ++mi) {
            const int rr = r0 + wr * 64 + mi * 16 + (lane >> 4) * 8;
            v8f c = acc[mi][ni];
#pragma unroll
            for (int i = 0; i < 8; ++i) {
                const float v = c[i] + bv;
                const float o = __shfl_xor(v, 1, 32);           // partner column value
                const int row = rr + i;
                const bool valid = (row >= estart) && (row < eend);
                if ((lane & 1) == 0) {                          // even lane = glu column
                    const float g = fminf(v, LIMIT);
                    const float l = fminf(fmaxf(o, -LIMIT), LIMIT);
                    const float s = 1.0f / (1.0f + __expf(-ALPHA * g));
                    const float r = valid ? (g * s * (l + 1.0f)) : 0.0f;
                    hb[(size_t)row * H + (col >> 1)] = (__bf16)r;
                }
            }
        }
    }
}

// ---------------- GEMM2 + bias, writes fp32 out[N,D] ----------------
__global__ __launch_bounds__(256)
void gemm2_kernel(const __bf16* __restrict__ hb,
                  const __bf16* __restrict__ w2b,
                  const float*  __restrict__ bias2,
                  const int*    __restrict__ ntpe,
                  float* __restrict__ out) {
    const int lane = threadIdx.x & 31;
    const int wave = threadIdx.x >> 5;
    const int wr = wave >> 1, wc = wave & 1;
    const int r0 = blockIdx.y * BM;
    const int c0 = blockIdx.x * BN;

    int estart, eend;
    const int e = expert_of(ntpe, r0, estart, eend);
    if (e < 0) {  // fully padded tile: still must write zeros to d_out
        for (int t = threadIdx.x; t < BM * BN; t += 256) {
            const int rrow = r0 + t / BN;
            const int ccol = c0 + t % BN;
            out[(size_t)rrow * D + ccol] = 0.0f;
        }
        return;
    }

    const __bf16* Ag = hb  + (size_t)r0 * H;
    const __bf16* Bg = w2b + ((size_t)e * D + c0) * (size_t)H;

    v8f acc[4][2];
#pragma unroll
    for (int mi = 0; mi < 4; ++mi)
#pragma unroll
        for (int ni = 0; ni < 2; ++ni) acc[mi][ni] = 0.0f;

    staged_mma(Ag, Bg, H, acc);

#pragma unroll
    for (int ni = 0; ni < 2; ++ni) {
        const int col = c0 + wc * 32 + ni * 16 + (lane & 15);
        const float bv = bias2[(size_t)e * D + col];
#pragma unroll
        for (int mi = 0; mi < 4; ++mi) {
            const int rr = r0 + wr * 64 + mi * 16 + (lane >> 4) * 8;
            v8f c = acc[mi][ni];
#pragma unroll
            for (int i = 0; i < 8; ++i) {
                const int row = rr + i;
                const bool valid = (row >= estart) && (row < eend);
                out[(size_t)row * D + col] = valid ? (c[i] + bv) : 0.0f;
            }
        }
    }
}

// ---------------- host-side launcher ----------------
extern "C" void kernel_launch(void* const* d_in, const int* in_sizes, int n_in,
                              void* d_out, int out_size, void* d_ws, size_t ws_size,
                              hipStream_t stream) {
    const float* x    = (const float*)d_in[0];   // [N, D]
    const float* w1   = (const float*)d_in[1];   // [E, 2H, D]
    const float* b1   = (const float*)d_in[2];   // [E, 2H]
    const float* w2   = (const float*)d_in[3];   // [E, D, H]
    const float* b2   = (const float*)d_in[4];   // [E, D]
    const int*   ntpe = (const int*)d_in[5];     // [E]

    // workspace layout (all offsets 256B-aligned; ~560 MB total)
    const size_t XB_BYTES  = (size_t)NTOK * D * 2;
    const size_t W1B_BYTES = (size_t)E * H2 * D * 2;
    const size_t W2B_BYTES = (size_t)E * D * H * 2;
    char* ws = (char*)d_ws;
    __bf16* xb  = (__bf16*)(ws);
    __bf16* w1b = (__bf16*)(ws + XB_BYTES);
    __bf16* w2b = (__bf16*)(ws + XB_BYTES + W1B_BYTES);
    __bf16* hb  = (__bf16*)(ws + XB_BYTES + W1B_BYTES + W2B_BYTES);  // [N, H] bf16

    const long nx  = (long)NTOK * D;
    const long nw1 = (long)E * H2 * D;
    const long nw2 = (long)E * D * H;

    cvt_f32_bf16_kernel<<<(int)((nx  + 2047) / 2048), 256, 0, stream>>>(x,  xb,  nx);
    cvt_f32_bf16_kernel<<<(int)((nw1 + 2047) / 2048), 256, 0, stream>>>(w1, w1b, nw1);
    cvt_f32_bf16_kernel<<<(int)((nw2 + 2047) / 2048), 256, 0, stream>>>(w2, w2b, nw2);

    gemm1_swiglu_kernel<<<dim3(H2 / BN, NTOK / BM), 256, 0, stream>>>(xb, w1b, b1, ntpe, hb);
    gemm2_kernel<<<dim3(D / BN, NTOK / BM), 256, 0, stream>>>(hb, w2b, b2, ntpe,
                                                              (float*)d_out);
}
